// WeightedCosine_85109071938344
// MI455X (gfx1250) — matrix operand order repaired and checked by reading, not compiled, not run on hardware
//
#include <hip/hip_runtime.h>
#include <hip/hip_bf16.h>

typedef __attribute__((ext_vector_type(16))) __bf16 bf16x16;
typedef __attribute__((ext_vector_type(8)))  __bf16 bf16x8;
typedef __attribute__((ext_vector_type(8)))  float  f32x8;

#define NN 8192
#define DD 512
#define PP 4
#define KK (PP * DD)   // 2048

// ---------------------------------------------------------------------------
// Kernel 1: F[n, p*DD + d] = (e[n,d] * w[p,d]) / (max(||e_n ⊙ w_p||2, eps) * sqrt(P))
// stored as bf16, row-major [NN, KK]. One block per row n; 256 threads cover
// D=512 as 2 consecutive elements each (packed 4-byte bf16x2 stores).
// ---------------------------------------------------------------------------
__global__ __launch_bounds__(256)
void wc_prep(const float* __restrict__ emb, const float* __restrict__ w,
             __bf16* __restrict__ F) {
  __shared__ float red[256];
  const int n   = blockIdx.x;
  const int tid = threadIdx.x;
  const int d0  = tid * 2;

  const float e0 = emb[(size_t)n * DD + d0];
  const float e1 = emb[(size_t)n * DD + d0 + 1];

  #pragma unroll
  for (int p = 0; p < PP; ++p) {
    const float v0 = e0 * w[p * DD + d0];
    const float v1 = e1 * w[p * DD + d0 + 1];
    red[tid] = v0 * v0 + v1 * v1;
    __syncthreads();
    #pragma unroll
    for (int s = 128; s > 0; s >>= 1) {
      if (tid < s) red[tid] += red[tid + s];
      __syncthreads();
    }
    const float sumsq = red[0];
    __syncthreads();  // red[] is reused next p iteration
    const float scale = 0.5f / fmaxf(sqrtf(sumsq), 1e-12f);  // 0.5 = 1/sqrt(P)

    struct bf2 { __bf16 x, y; };
    bf2 pkt;
    pkt.x = (__bf16)(v0 * scale);
    pkt.y = (__bf16)(v1 * scale);
    *reinterpret_cast<bf2*>(F + (size_t)n * KK + p * DD + d0) = pkt;
  }
}

// ---------------------------------------------------------------------------
// Kernel 2: attention = F * F^T (symmetric SYRK), bf16 WMMA, f32 accumulate.
// Block = 128 threads = 4 waves; block tile 128x128; wave tile 64x64
// (4x4 grid of v_wmma_f32_16x16x32_bf16 accumulators).
//
// Per-lane fragment addressing follows the CDNA5 ISA VGPR layouts:
//   A (16x32 bf16):  lane<16: row rowBase+lane, K = {k0..k0+7, k0+16..k0+23}
//                    lane>=16: row rowBase+lane-16, K shifted by +8
//   B (32x16 bf16):  lane<16: row colBase+lane (B column == F row), K = {k0..k0+7, k0+16..k0+23}
//                    lane>=16: same rows, K shifted by +8
// All fragments are contiguous 16-byte loads from row-major F (L2-resident).
// Only upper block-triangle computed; off-diagonal tiles mirrored on store.
// ---------------------------------------------------------------------------
__global__ __launch_bounds__(128)
void wc_gemm(const __bf16* __restrict__ F, float* __restrict__ out) {
  const int bx = blockIdx.x;   // column block
  const int by = blockIdx.y;   // row block
  if (bx < by) return;         // symmetry: skip lower triangle

  const int lane = threadIdx.x;   // 0..31 (wave32)
  const int wave = threadIdx.y;   // 0..3
  const int half = lane >> 4;     // 0 | 1 (half-wave)
  const int l16  = lane & 15;

  const int rowBlk = by * 128 + (wave >> 1) * 64;
  const int colBlk = bx * 128 + (wave &  1) * 64;

  f32x8 c[4][4];
  #pragma unroll
  for (int m = 0; m < 4; ++m)
    #pragma unroll
    for (int nn = 0; nn < 4; ++nn)
      c[m][nn] = f32x8{0.f, 0.f, 0.f, 0.f, 0.f, 0.f, 0.f, 0.f};

  for (int k0 = 0; k0 < KK; k0 += 32) {
    bf16x16 a[4];
    #pragma unroll
    for (int m = 0; m < 4; ++m) {
      const __bf16* p = F + (size_t)(rowBlk + m * 16 + l16) * KK + k0 + half * 8;
      bf16x8 lo = *reinterpret_cast<const bf16x8*>(p);        // K k0(+8)..
      bf16x8 hi = *reinterpret_cast<const bf16x8*>(p + 16);   // K k0+16(+8)..
      a[m] = __builtin_shufflevector(lo, hi, 0,1,2,3,4,5,6,7,8,9,10,11,12,13,14,15);
    }
    #pragma unroll
    for (int nn = 0; nn < 4; ++nn) {
      const __bf16* p = F + (size_t)(colBlk + nn * 16 + l16) * KK + k0 + half * 8;
      bf16x8 lo = *reinterpret_cast<const bf16x8*>(p);        // K k0(+8)..
      bf16x8 hi = *reinterpret_cast<const bf16x8*>(p + 16);   // K k0+16(+8)..
      bf16x16 b = __builtin_shufflevector(lo, hi, 0,1,2,3,4,5,6,7,8,9,10,11,12,13,14,15);
      #pragma unroll
      for (int m = 0; m < 4; ++m)
        c[m][nn] = __builtin_amdgcn_wmma_f32_16x16x32_bf16(
            /*neg_a=*/false, a[m], /*neg_b=*/false, b,
            /*c_mod=*/(short)0, c[m][nn], /*reuse_a=*/false, /*reuse_b=*/false);
    }
  }

  // C/D layout: lane<16 -> N=l16, VGPR r -> M=r ; lane>=16 -> M=8+r
  const bool mirror = (bx != by);
  #pragma unroll
  for (int m = 0; m < 4; ++m) {
    const int row0 = rowBlk + m * 16 + half * 8;
    #pragma unroll
    for (int nn = 0; nn < 4; ++nn) {
      const int col = colBlk + nn * 16 + l16;
      #pragma unroll
      for (int r = 0; r < 8; ++r) {
        const float val = c[m][nn][r];
        out[(size_t)(row0 + r) * NN + col] = val;
        if (mirror)
          out[(size_t)col * NN + (row0 + r)] = val;  // symmetric mirror
      }
    }
  }
}

// ---------------------------------------------------------------------------
extern "C" void kernel_launch(void* const* d_in, const int* in_sizes, int n_in,
                              void* d_out, int out_size, void* d_ws, size_t ws_size,
                              hipStream_t stream) {
  (void)in_sizes; (void)n_in; (void)out_size; (void)ws_size;
  const float* emb = (const float*)d_in[0];   // [8192, 512] fp32
  const float* w   = (const float*)d_in[1];   // [4, 512]    fp32
  float* out = (float*)d_out;                 // [8192, 8192] fp32
  __bf16* F = (__bf16*)d_ws;                  // [8192, 2048] bf16 = 32 MB scratch

  wc_prep<<<dim3(NN), dim3(256), 0, stream>>>(emb, w, F);
  wc_gemm<<<dim3(NN / 128, NN / 128), dim3(32, 4), 0, stream>>>(F, out);
}
